// EfficientInteractionBilinear_10359461118182
// MI455X (gfx1250) — matrix operand order repaired and checked by reading, not compiled, not run on hardware
//
#include <hip/hip_runtime.h>
#include <hip/hip_bf16.h>

// ---------------------------------------------------------------------------
// EfficientInteractionBilinear, fused, bf16-WMMA (gfx1250 / MI455X)
//
// Shapes: E=50000, S=16, KMAX=16, K_AVG=8, EMB=128, INTERM=64, UNITS_OUT=128
// out[e,o] = sum_{i,c,s,k} rbf[e,i,s] * sph[e,s,k] * m2[e,k,c] * W[c,i,o]
//
// Roofline: ~490MB traffic => ~21us @23.3TB/s; 121 GFLOP.  f32 matrix path
// ~200us => bf16 WMMA w/ f32 accumulation => ~24us, balanced vs HBM.
// B-operand LDS rows are zero-padded to K=32 (no masking valu), the phase-3
// A-panel uses a block-permuted layout so each lane's A slice is ONE 32B
// ds load, and all bf16 packing uses native __bf16 conversions.
// LDS: 296KB of the 320KB WGP budget (1 WG/WGP by design; compute-heavy).
// ---------------------------------------------------------------------------

typedef __attribute__((ext_vector_type(16))) __bf16          v16bf;
typedef __attribute__((ext_vector_type(8)))  __bf16          v8bf;
typedef __attribute__((ext_vector_type(4)))  __bf16          v4bf;
typedef __attribute__((ext_vector_type(8)))  float           v8f;
typedef __attribute__((ext_vector_type(4)))  float           v4f;
typedef __attribute__((ext_vector_type(16))) unsigned short  v16u16;
typedef __attribute__((ext_vector_type(8)))  unsigned short  v8u16;
typedef __attribute__((ext_vector_type(4)))  unsigned short  v4u16;

#define E_EDGES   50000
#define S_DIM     16
#define KMAX      16
#define K_AVG     8
#define EMB       128
#define INTERM    64
#define OUT_DIM   128
#define BE        16               // edges per workgroup
#define NBLK      (E_EDGES / BE)   // 3125
#define NCHUNK    8                // 128 channels / 16
#define KSTEPS    32               // 1024 k' per chunk / 32
#define KPAD      32               // padded K extent of B-operand rows

static __device__ __forceinline__ unsigned short f2bf(float f) {
    return __builtin_bit_cast(unsigned short, (__bf16)f);
}

static __device__ __forceinline__ v8f wmma_bf16(v16u16 au, v16u16 bu, v8f c) {
    v16bf a = __builtin_bit_cast(v16bf, au);
    v16bf b = __builtin_bit_cast(v16bf, bu);
    return __builtin_amdgcn_wmma_f32_16x16x32_bf16(
        /*neg_a=*/false, a, /*neg_b=*/false, b,
        /*c_mod=*/(short)0, c, /*reuse_a=*/false, /*reuse_b=*/false);
}

// ---------------------------------------------------------------------------
// Kernel 1: repack weight (EMB,INTERM,OUT) f32 -> bf16 B-operand tiles.
// Layout: [chunk(8)][kstep(32)][ntile(8)][lane(32)][j(16)].  For lane L
// (g=L>>4, col=L&15): element j covers K = k' = ks*32 + 8g + j (j<8),
// ks*32 + 16 + 8g + (j-8) (j>=8),
// N = o = ntile*16 + col.  k' is C-MAJOR: cidx = k'>>6, i = k'&63.
// ---------------------------------------------------------------------------
__global__ __launch_bounds__(256) void repack_weight_kernel(
    const float* __restrict__ w, unsigned short* __restrict__ wp) {
    int tid = blockIdx.x * 256 + threadIdx.x;      // 0 .. 65535
    int L  = tid & 31;
    int nt = (tid >> 5) & 7;
    int ks = (tid >> 8) & 31;
    int ch = (tid >> 13) & 7;
    int g  = L >> 4;
    int o  = nt * 16 + (L & 15);
    unsigned short vals[16];
#pragma unroll
    for (int j = 0; j < 16; ++j) {
        int kp   = ks * 32 + (j < 8 ? 8 * g + j : 16 + 8 * g + (j - 8));
        int i    = kp & 63;                 // c-major k'
        int c    = ch * 16 + (kp >> 6);
        vals[j]  = f2bf(w[(c * INTERM + i) * OUT_DIM + o]);
    }
    unsigned short* dst = wp + (size_t)tid * 16;
#pragma unroll
    for (int j = 0; j < 16; ++j) dst[j] = vals[j];
}

// ---------------------------------------------------------------------------
// Kernel 2: fused interaction. 1 WG = 16 edges, 8 waves (256 thr, wave32).
// Dynamic LDS carve (ushorts):
//   sm_m2   [e][c][32k, 16..31 zero] 16*128*32 (128KB) -- reused as sm_A
//   sm_sumk [e][c][32s, 16..31 zero] 16*128*32 (128KB)
//   sm_rbf  [e][i][16s]              16*64*16  ( 32KB)
//   sm_sph  [e][s][16k]              16*16*16  (  8KB)
// total 296KB of the 320KB WGP LDS.
// ---------------------------------------------------------------------------
extern "C" __global__ __launch_bounds__(256) void eib_fused_kernel(
    const float* __restrict__ rbf_W1,      // (E,64,16)
    const float* __restrict__ sph,         // (E,16,16)
    const float* __restrict__ m,           // (T,128)
    const int*   __restrict__ id_reduce,   // (T)
    const int*   __restrict__ id_ragged,   // (T)
    const unsigned short* __restrict__ wpack,  // packed bf16 weight
    float* __restrict__ out)               // (E,128)
{
    extern __shared__ unsigned short smem[];
    unsigned short* sm_m2   = smem;                        // 65536
    unsigned short* sm_sumk = sm_m2 + BE * EMB * KPAD;     // 65536
    unsigned short* sm_rbf  = sm_sumk + BE * EMB * KPAD;   // 16384
    unsigned short* sm_sph  = sm_rbf + BE * INTERM * S_DIM;//  4096
    unsigned short* sm_A    = sm_m2;                       // 16*1024 (reuse)

    const int tid  = threadIdx.x;
    const int wave = tid >> 5;
    const int L    = tid & 31;
    const int g    = L >> 4;
    const int r    = L & 15;
    const int blk  = blockIdx.x;
    const int eb   = blk * BE;

    // ---- Phase 0a: zero m2 + sumk (256KB) with b128 stores --------------
    {
        uint4 z = {0u, 0u, 0u, 0u};
        uint4* zp = (uint4*)smem;          // covers sm_m2 + sm_sumk
#pragma unroll
        for (int q = 0; q < 64; ++q) zp[tid + q * 256] = z;
    }
    __syncthreads();

    // ---- Phase 0b: ragged gather of m (into [e][c][32k]) + stage sph/rbf
    {
        // 128 triplets for this block; 2 threads per triplet (64 chans each)
        int tt   = tid >> 1;
        int part = tid & 1;
        long t   = (long)blk * (BE * K_AVG) + tt;
        int er   = id_reduce[t];
        int kr   = id_ragged[t];
        if (er >= eb && er < eb + BE && kr >= 0 && kr < KMAX) {
            unsigned short* dst =
                sm_m2 + ((er - eb) * EMB + part * 64) * KPAD + kr;
            const v4f* src = (const v4f*)(m + (size_t)t * EMB + part * 64);
#pragma unroll
            for (int q = 0; q < 16; ++q) {
                v4f v = src[q];
                dst[(q * 4 + 0) * KPAD] = f2bf(v.x);
                dst[(q * 4 + 1) * KPAD] = f2bf(v.y);
                dst[(q * 4 + 2) * KPAD] = f2bf(v.z);
                dst[(q * 4 + 3) * KPAD] = f2bf(v.w);
            }
        }
        // sph: 4096 f32 -> bf16, vectorized 4-wide
        const v4f* sphv = (const v4f*)(sph + (size_t)eb * S_DIM * KMAX);
#pragma unroll
        for (int q = 0; q < 4; ++q) {
            int idx = tid * 4 + q * 1024;
            v4bf b = __builtin_convertvector(sphv[idx >> 2], v4bf);
            *(v4u16*)(sm_sph + idx) = __builtin_bit_cast(v4u16, b);
        }
        // rbf: 16384 f32 -> bf16, vectorized 4-wide
        const v4f* rbfv = (const v4f*)(rbf_W1 + (size_t)eb * INTERM * S_DIM);
#pragma unroll
        for (int q = 0; q < 16; ++q) {
            int idx = tid * 4 + q * 1024;
            v4bf b = __builtin_convertvector(rbfv[idx >> 2], v4bf);
            *(v4u16*)(sm_rbf + idx) = __builtin_bit_cast(v4u16, b);
        }
    }
    __syncthreads();

    // ---- Phase 1: sum_k[e] = sph[e](16x16) @ m2[e](16x128), K pad 16->32
    // 128 (edge,ntile) jobs, 16 per wave.
#pragma unroll
    for (int jj = 0; jj < 16; ++jj) {
        int job = wave * 16 + jj;
        int e   = job >> 3;
        int nt  = job & 7;
        // A: row M=s=r ; low K run {8g..8g+7}; K>=16 is the zero pad
        v8u16 alo = *(const v8u16*)(sm_sph + (e * S_DIM + r) * KMAX + 8 * g);
        v16u16 au;
#pragma unroll
        for (int j = 0; j < 8; ++j) { au[j] = alo[j]; au[j + 8] = 0; }
        // B: column N=c=nt*16+r ; K runs {8g..8g+7} U {16+8g..} (pad)
        v8u16 blo = *(const v8u16*)(sm_m2 + (e * EMB + nt * 16 + r) * KPAD
                                    + 8 * g);
        v8u16 bhi = *(const v8u16*)(sm_m2 + (e * EMB + nt * 16 + r) * KPAD
                                    + 16 + 8 * g);
        v16u16 bu;
#pragma unroll
        for (int j = 0; j < 8; ++j) { bu[j] = blo[j]; bu[j + 8] = bhi[j]; }
        v8f c = {};
        v8f d = wmma_bf16(au, bu, c);
        // D element j -> (s=j+8g, c=nt*16+r): contiguous run in [e][c][32s]
        v8bf st = __builtin_convertvector(d, v8bf);
        *(v8u16*)(sm_sumk + (e * EMB + nt * 16 + r) * KPAD + 8 * g) =
            __builtin_bit_cast(v8u16, st);
    }
    __syncthreads();

    // persistent accumulator: this wave owns output tile o = wave*16..+15
    v8f acc = {};

    // ---- chunk loop over 16-wide c slices -------------------------------
    for (int ch = 0; ch < NCHUNK; ++ch) {
        // Phase 2: A-panel[e][k'=c*64+i] = rbf[e](64x16) @ sumk[e][:,chunk]
        // 64 (edge,itile) jobs, 8 per wave. K=s padded 16->32.
#pragma unroll
        for (int jj = 0; jj < 8; ++jj) {
            int job = wave * 8 + jj;
            int e   = job >> 2;
            int it  = job & 3;
            v8u16 alo = *(const v8u16*)(sm_rbf +
                            (e * INTERM + it * 16 + r) * S_DIM + 8 * g);
            v16u16 au;
#pragma unroll
            for (int j = 0; j < 8; ++j) { au[j] = alo[j]; au[j + 8] = 0; }
            v8u16 blo = *(const v8u16*)(sm_sumk +
                            (e * EMB + ch * 16 + r) * KPAD + 8 * g);
            v8u16 bhi = *(const v8u16*)(sm_sumk +
                            (e * EMB + ch * 16 + r) * KPAD + 16 + 8 * g);
            v16u16 bu;
#pragma unroll
            for (int j = 0; j < 8; ++j) { bu[j] = blo[j]; bu[j + 8] = bhi[j]; }
            v8f c = {};
            v8f d = wmma_bf16(au, bu, c);
            // D element j -> (i = it*16+j+8g, c = r): k' = r*64 + i.
            // Store with the middle-8-block swap inside each 32-k' group so
            // phase-3 lanes read ONE contiguous 32B slice:
            //   addr = r*64 + (it>>1)*32 + g*16 + (it&1)*8
            v8bf st = __builtin_convertvector(d, v8bf);
            *(v8u16*)(sm_A + e * 1024 + r * 64 + (it >> 1) * 32
                      + g * 16 + (it & 1) * 8) =
                __builtin_bit_cast(v8u16, st);
        }
        __syncthreads();

        // Phase 3: out-tile GEMM  A(16e x 1024) @ Wpack(1024 x 128)
        // wave -> ntile=wave ; 32 K-steps ; permuted A-panel => each lane's
        // documented slice {8g..8g+7}U{16+8g..} is one 32B ds load.
        const v16u16* wp = (const v16u16*)wpack;
        size_t wbase = ((size_t)(ch * KSTEPS) * 8 + wave) * 32 + L;
        for (int ks = 0; ks < KSTEPS; ++ks) {
            v16u16 au = *(const v16u16*)(sm_A + r * 1024 + ks * 32 + 16 * g);
            v16u16 bu = wp[wbase + (size_t)ks * 8 * 32];
            acc = wmma_bf16(au, bu, acc);
        }
        __syncthreads();   // protect sm_A before next chunk overwrites it
    }

    // ---- epilogue: D rows are edges, cols are o = wave*16 + r -----------
#pragma unroll
    for (int j = 0; j < 8; ++j) {
        int e = j + 8 * g;
        out[(size_t)(eb + e) * OUT_DIM + wave * 16 + r] = acc[j];
    }
}

// ---------------------------------------------------------------------------
extern "C" void kernel_launch(void* const* d_in, const int* in_sizes, int n_in,
                              void* d_out, int out_size, void* d_ws, size_t ws_size,
                              hipStream_t stream) {
    const float* rbf_W1    = (const float*)d_in[0];   // (E,64,16)
    const float* sph       = (const float*)d_in[1];   // (E,16,16)
    const float* m         = (const float*)d_in[2];   // (T,128)
    const float* weight    = (const float*)d_in[3];   // (128,64,128)
    const int*   id_reduce = (const int*)d_in[4];     // (T)
    const int*   id_ragged = (const int*)d_in[5];     // (T)
    float*       out       = (float*)d_out;

    unsigned short* wpack = (unsigned short*)d_ws;    // 2MB of workspace

    // 1M packed bf16 elements, 16 per thread -> 65536 threads
    repack_weight_kernel<<<256, 256, 0, stream>>>(weight, wpack);

    const size_t lds_bytes =
        (size_t)(BE * EMB * KPAD + BE * EMB * KPAD +
                 BE * INTERM * S_DIM + BE * S_DIM * KMAX)
        * sizeof(unsigned short);   // 303104
    eib_fused_kernel<<<NBLK, 256, lds_bytes, stream>>>(
        rbf_W1, sph, m, id_reduce, id_ragged, wpack, out);
}
